// MoELayer_72559177498897
// MI455X (gfx1250) — compile-verified, matrix operand order checked by reading, not run on hardware
//
#include <hip/hip_runtime.h>
#include <hip/hip_bf16.h>
#include <math.h>

#define HIDDEN 768
#define INTER  1536
#define NEXP   8

typedef __attribute__((ext_vector_type(16))) __bf16 v16bf;
typedef __attribute__((ext_vector_type(8)))  float  v8f;

__device__ __forceinline__ __bf16 f2bf(float f) { return (__bf16)f; }

// fast silu: g * sigmoid(g) with v_rcp_f32 (result feeds bf16 anyway)
__device__ __forceinline__ float fast_silu(float g) {
  return g * __builtin_amdgcn_rcpf(1.f + __expf(-g));
}

// ---------------- init: zero counters ----------------
__global__ __launch_bounds__(64) void moe_init(int* counts, float* psum) {
  int i = threadIdx.x;
  if (i < NEXP) { counts[i] = 0; psum[i] = 0.f; }
}

// ---------------- transpose + f32->bf16 convert: in [E][R][C] -> out [E][C][R] ----------------
__global__ __launch_bounds__(256) void moe_transpose(
    const float* __restrict__ in, __bf16* __restrict__ out, int R, int C)
{
  __shared__ float TSH[64][65];
  int e  = blockIdx.z;
  int r0 = blockIdx.y * 64, c0 = blockIdx.x * 64;
  const float* inE = in + (size_t)e * R * C;
  __bf16* outE = out + (size_t)e * R * C;
  int tid = threadIdx.x;
  for (int i = tid; i < 64 * 64; i += 256) {     // coalesced read along C
    int r = i >> 6, c = i & 63;
    TSH[r][c] = inE[(size_t)(r0 + r) * C + (c0 + c)];
  }
  __syncthreads();
  for (int i = tid; i < 64 * 64; i += 256) {     // coalesced write along R
    int c = i >> 6, r = i & 63;
    outE[(size_t)(c0 + c) * R + (r0 + r)] = f2bf(TSH[r][c]);  // 65-pad: conflict-free column read
  }
}

// ---------------- router: logits, softmax, top2, gather lists ----------------
__global__ __launch_bounds__(256) void moe_router(
    const float* __restrict__ x, const float* __restrict__ rw,
    int* __restrict__ counts, float* __restrict__ psum,
    int* __restrict__ eList, float* __restrict__ wtk, int T)
{
  __shared__ float RW[HIDDEN * NEXP];   // 24KB
  __shared__ float ps[NEXP];
  int tid = threadIdx.x;
  int t = blockIdx.x * 256 + tid;
  for (int i = tid; i < HIDDEN * NEXP; i += 256) RW[i] = rw[i];
  if (tid < NEXP) ps[tid] = 0.f;
  __syncthreads();

  if (t < T) {
    float acc[NEXP];
#pragma unroll
    for (int e = 0; e < NEXP; ++e) acc[e] = 0.f;
    const float4* xr4 = (const float4*)(x + (size_t)t * HIDDEN);
    for (int h4 = 0; h4 < HIDDEN / 4; ++h4) {
      float4 xv = xr4[h4];
      const float* rwp = &RW[h4 * 4 * NEXP];
#pragma unroll
      for (int e = 0; e < NEXP; ++e)
        acc[e] += xv.x * rwp[e] + xv.y * rwp[NEXP + e] +
                  xv.z * rwp[2 * NEXP + e] + xv.w * rwp[3 * NEXP + e];
    }
    float m = acc[0];
#pragma unroll
    for (int e = 1; e < NEXP; ++e) m = fmaxf(m, acc[e]);
    float p[NEXP]; float s = 0.f;
#pragma unroll
    for (int e = 0; e < NEXP; ++e) { p[e] = __expf(acc[e] - m); s += p[e]; }
    float inv = 1.f / s;
#pragma unroll
    for (int e = 0; e < NEXP; ++e) p[e] *= inv;

    int i0 = 0; float p0 = p[0];
#pragma unroll
    for (int e = 1; e < NEXP; ++e) if (p[e] > p0) { p0 = p[e]; i0 = e; }
    int i1 = (i0 == 0) ? 1 : 0; float p1 = p[i1];
#pragma unroll
    for (int e = 0; e < NEXP; ++e) if (e != i0 && p[e] > p1) { p1 = p[e]; i1 = e; }

    float dn = 1.f / (p0 + p1);
    wtk[t * 2 + 0] = p0 * dn;
    wtk[t * 2 + 1] = p1 * dn;
    int s0 = atomicAdd(&counts[i0], 1); eList[i0 * T + s0] = t * 2;
    int s1 = atomicAdd(&counts[i1], 1); eList[i1 * T + s1] = t * 2 + 1;
#pragma unroll
    for (int e = 0; e < NEXP; ++e) atomicAdd(&ps[e], p[e]);
  }
  __syncthreads();
  if (tid < NEXP) atomicAdd(&psum[tid], ps[tid]);
}

// ---------------- aux loss scalar ----------------
__global__ __launch_bounds__(64) void moe_aux(
    const int* __restrict__ counts, const float* __restrict__ psum,
    float* __restrict__ out_aux, int T)
{
  if (threadIdx.x == 0 && blockIdx.x == 0) {
    float s = 0.f;
    for (int e = 0; e < NEXP; ++e)
      s += ((float)counts[e] / (float)(T * 2)) * (psum[e] / (float)T);
    *out_aux = (float)NEXP * s * 0.01f;
  }
}

// ---------------- gate/up GEMMs + SiLU: M=32 tile, bf16 WMMA, f32 accum ----------------
// wgT/wuT layout: [E][INTER][HIDDEN] (N-major) so each lane's B chunk is contiguous.
__global__ __launch_bounds__(256) void moe_gateup(
    const float* __restrict__ x, const __bf16* __restrict__ wgT,
    const __bf16* __restrict__ wuT, const int* __restrict__ counts,
    const int* __restrict__ eList, __bf16* __restrict__ hb, int T)
{
  __shared__ __bf16 XA[32][HIDDEN];   // 48KB: 32 tokens x 768 bf16
  __shared__ int toks[32];
  int e   = blockIdx.y;
  int cnt = counts[e];
  int m0  = blockIdx.x * 32;
  if (m0 >= cnt) return;
  int tid   = threadIdx.x;
  int dummy = 2 * T;                  // pad rows target the dummy slot -> branch-free stores
  if (tid < 32) {
    int s = m0 + tid;
    toks[tid] = (s < cnt) ? eList[e * T + s] : dummy;
  }
  __syncthreads();
  for (int i = tid; i < 32 * HIDDEN; i += 256) {
    int r = i / HIDDEN, c = i % HIDDEN;
    int tk2 = toks[r];
    XA[r][c] = f2bf(tk2 < dummy ? x[(size_t)(tk2 >> 1) * HIDDEN + c] : 0.f);
  }
  __syncthreads();

  int wave = tid >> 5, lane = tid & 31;
  int hrow = lane & 15;               // A: M row (per half) / B: N column per lane
  int ksel = (lane >> 4) * 8;         // K sub-chunk per lane half
  const __bf16* wgE = wgT + (size_t)e * HIDDEN * INTER;
  const __bf16* wuE = wuT + (size_t)e * HIDDEN * INTER;

  for (int nt = wave; nt < INTER / 16; nt += 8) {   // 12 N-tiles per wave
    int nb = nt * 16;
    const __bf16* wgRow = wgE + (size_t)(nb + hrow) * HIDDEN;
    const __bf16* wuRow = wuE + (size_t)(nb + hrow) * HIDDEN;
    v8f cg0 = {}, cg1 = {}, cu0 = {}, cu1 = {};
    for (int kb = 0; kb < HIDDEN; kb += 32) {       // 24 K-steps
      v16bf a0, a1, bg, bu;
#pragma unroll
      for (int j = 0; j < 8; ++j) {
        a0[j]     = XA[hrow][kb + ksel + j];
        a0[j + 8] = XA[hrow][kb + 16 + ksel + j];
        a1[j]     = XA[16 + hrow][kb + ksel + j];
        a1[j + 8] = XA[16 + hrow][kb + 16 + ksel + j];
        bg[j]     = wgRow[kb + ksel + j];
        bg[j + 8] = wgRow[kb + 16 + ksel + j];
        bu[j]     = wuRow[kb + ksel + j];
        bu[j + 8] = wuRow[kb + 16 + ksel + j];
      }
      cg0 = __builtin_amdgcn_wmma_f32_16x16x32_bf16(false, a0, false, bg, (short)0, cg0, false, false);
      cg1 = __builtin_amdgcn_wmma_f32_16x16x32_bf16(false, a1, false, bg, (short)0, cg1, false, false);
      cu0 = __builtin_amdgcn_wmma_f32_16x16x32_bf16(false, a0, false, bu, (short)0, cu0, false, false);
      cu1 = __builtin_amdgcn_wmma_f32_16x16x32_bf16(false, a1, false, bu, (short)0, cu1, false, false);
    }
    // D layout: VGPR i, lane l -> M = i + 8*(l>=16), N = l&15
    int ncol = nb + (lane & 15);
    int mb   = (lane >> 4) * 8;
#pragma unroll
    for (int i = 0; i < 8; ++i) {
      float h0 = fast_silu(cg0[i]) * cu0[i];           // branch-free, rcp-based
      hb[(size_t)toks[mb + i] * INTER + ncol] = f2bf(h0);
      float h1 = fast_silu(cg1[i]) * cu1[i];
      hb[(size_t)toks[16 + mb + i] * INTER + ncol] = f2bf(h1);
    }
  }
}

// ---------------- down GEMM: M=32 tile, K staged in two 768-wide halves ----------------
// wdT layout: [E][HIDDEN][INTER] (N-major). A-tile staging uses async global->LDS copies.
__global__ __launch_bounds__(256) void moe_down(
    const __bf16* __restrict__ hb, const __bf16* __restrict__ wdT,
    const int* __restrict__ counts, const int* __restrict__ eList,
    float* __restrict__ od, int T)
{
  __shared__ __bf16 HA[32][INTER / 2];   // 48KB: 32 slots x 768 bf16 (half of K)
  __shared__ int toks[32];
  int e   = blockIdx.y;
  int cnt = counts[e];
  int m0  = blockIdx.x * 32;
  if (m0 >= cnt) return;
  int tid   = threadIdx.x;
  int dummy = 2 * T;
  if (tid < 32) {
    int s = m0 + tid;
    toks[tid] = (s < cnt) ? eList[e * T + s] : dummy;
  }
  __syncthreads();

  int wave = tid >> 5, lane = tid & 31;
  int hrow = lane & 15;
  int ksel = (lane >> 4) * 8;
  const __bf16* wdE = wdT + (size_t)e * INTER * HIDDEN;

  v8f acc0[6], acc1[6];                 // 6 N-tiles per wave (48/8), 2 M halves
#pragma unroll
  for (int ni = 0; ni < 6; ++ni) { acc0[ni] = (v8f){}; acc1[ni] = (v8f){}; }

  for (int kp = 0; kp < 2; ++kp) {
    int kbase = kp * (INTER / 2);
    // async bf16 copy hb rows -> LDS: 16B per lane, no VGPR round-trip (ASYNCcnt-tracked)
    for (int i = tid; i < 32 * (INTER / 2) / 8; i += 256) {   // 3072 16B chunks
      int r = i / 96, c = (i % 96) * 8;                       // 96 chunks per row
      const __bf16* src = hb + (size_t)toks[r] * INTER + kbase + c;
      unsigned lofs = (unsigned)(uintptr_t)(&HA[r][c]);
      asm volatile("global_load_async_to_lds_b128 %0, %1, off"
                   :: "v"(lofs), "v"(src) : "memory");
    }
    asm volatile("s_wait_asynccnt 0" ::: "memory");
    __syncthreads();
#pragma unroll
    for (int ni = 0; ni < 6; ++ni) {
      int nb = (wave + ni * 8) * 16;
      const __bf16* wRow = wdE + (size_t)(nb + hrow) * INTER + kbase;
      for (int kb = 0; kb < INTER / 2; kb += 32) {    // 24 K-steps per half
        v16bf a0, a1, b;
#pragma unroll
        for (int j = 0; j < 8; ++j) {
          a0[j]     = HA[hrow][kb + ksel + j];
          a0[j + 8] = HA[hrow][kb + 16 + ksel + j];
          a1[j]     = HA[16 + hrow][kb + ksel + j];
          a1[j + 8] = HA[16 + hrow][kb + 16 + ksel + j];
          b[j]      = wRow[kb + ksel + j];
          b[j + 8]  = wRow[kb + 16 + ksel + j];
        }
        acc0[ni] = __builtin_amdgcn_wmma_f32_16x16x32_bf16(false, a0, false, b, (short)0, acc0[ni], false, false);
        acc1[ni] = __builtin_amdgcn_wmma_f32_16x16x32_bf16(false, a1, false, b, (short)0, acc1[ni], false, false);
      }
    }
    __syncthreads();
  }

  int mb = (lane >> 4) * 8;
#pragma unroll
  for (int ni = 0; ni < 6; ++ni) {
    int ncol = (wave + ni * 8) * 16 + (lane & 15);
#pragma unroll
    for (int i = 0; i < 8; ++i) {
      od[(size_t)toks[mb + i] * HIDDEN + ncol]      = acc0[ni][i];  // unique slot: no atomics
      od[(size_t)toks[16 + mb + i] * HIDDEN + ncol] = acc1[ni][i];
    }
  }
}

// ---------------- combine the two expert contributions per token ----------------
__global__ __launch_bounds__(256) void moe_combine(
    const float* __restrict__ od, const float* __restrict__ wtk,
    float* __restrict__ out, int T)
{
  int idx = blockIdx.x * 256 + threadIdx.x;
  int total = T * HIDDEN / 4;
  if (idx >= total) return;
  int t = (idx * 4) / HIDDEN;
  int c = (idx * 4) % HIDDEN;
  float4 a = *(const float4*)(od + (size_t)(t * 2) * HIDDEN + c);
  float4 b = *(const float4*)(od + (size_t)(t * 2 + 1) * HIDDEN + c);
  float w0 = wtk[t * 2], w1 = wtk[t * 2 + 1];
  float4 r;
  r.x = w0 * a.x + w1 * b.x;
  r.y = w0 * a.y + w1 * b.y;
  r.z = w0 * a.z + w1 * b.z;
  r.w = w0 * a.w + w1 * b.w;
  *(float4*)(out + (size_t)t * HIDDEN + c) = r;
}

extern "C" void kernel_launch(void* const* d_in, const int* in_sizes, int n_in,
                              void* d_out, int out_size, void* d_ws, size_t ws_size,
                              hipStream_t stream) {
  const float* x  = (const float*)d_in[0];
  const float* rw = (const float*)d_in[1];
  const float* wg = (const float*)d_in[2];
  const float* wu = (const float*)d_in[3];
  const float* wd = (const float*)d_in[4];
  float* out = (float*)d_out;
  int T = in_sizes[0] / HIDDEN;   // 2048

  // workspace layout (~82 MB)
  char* ws = (char*)d_ws;
  size_t off = 0;
  int*    counts = (int*)(ws + off);    off += 256;
  float*  psum   = (float*)(ws + off);  off += 256;
  int*    eList  = (int*)(ws + off);    off += (size_t)NEXP * T * sizeof(int);
  off = (off + 255) & ~(size_t)255;
  float*  wtk    = (float*)(ws + off);  off += (size_t)T * 2 * sizeof(float);
  off = (off + 255) & ~(size_t)255;
  __bf16* hb     = (__bf16*)(ws + off); off += (size_t)(T * 2 + 1) * INTER * sizeof(unsigned short);
  off = (off + 255) & ~(size_t)255;
  float*  od     = (float*)(ws + off);  off += (size_t)(T * 2 + 1) * HIDDEN * sizeof(float);
  off = (off + 255) & ~(size_t)255;
  const size_t wsz = (size_t)NEXP * HIDDEN * INTER;
  __bf16* wgT    = (__bf16*)(ws + off); off += wsz * sizeof(unsigned short);
  __bf16* wuT    = (__bf16*)(ws + off); off += wsz * sizeof(unsigned short);
  __bf16* wdT    = (__bf16*)(ws + off); off += wsz * sizeof(unsigned short);
  (void)ws_size; (void)n_in; (void)out_size;

  moe_init<<<1, 64, 0, stream>>>(counts, psum);
  moe_router<<<(T + 255) / 256, 256, 0, stream>>>(x, rw, counts, psum, eList, wtk, T);
  moe_aux<<<1, 64, 0, stream>>>(counts, psum, out + (size_t)T * HIDDEN, T);

  // one-time weight transpose + bf16 convert (N-major layouts for contiguous B chunks)
  dim3 gtGU(INTER / 64, HIDDEN / 64, NEXP);
  moe_transpose<<<gtGU, 256, 0, stream>>>(wg, wgT, HIDDEN, INTER);
  moe_transpose<<<gtGU, 256, 0, stream>>>(wu, wuT, HIDDEN, INTER);
  dim3 gtD(HIDDEN / 64, INTER / 64, NEXP);
  moe_transpose<<<gtD, 256, 0, stream>>>(wd, wdT, INTER, HIDDEN);

  dim3 gridE((T + 31) / 32, NEXP);
  moe_gateup<<<gridE, 256, 0, stream>>>(x, wgT, wuT, counts, eList, hb, T);
  moe_down  <<<gridE, 256, 0, stream>>>(hb, wdT, counts, eList, od, T);
  moe_combine<<<(T * HIDDEN / 4 + 255) / 256, 256, 0, stream>>>(od, wtk, out, T);
}
